// BLSTMCRF_52269751992402
// MI455X (gfx1250) — matrix operand order repaired and checked by reading, not compiled, not run on hardware
//
#include <hip/hip_runtime.h>
#include <hip/hip_bf16.h>
#include <stdint.h>

#define TT 512
#define BB 32
#define DD 192
#define HH 128
#define G4H 512
#define LL 32

typedef __attribute__((ext_vector_type(16))) __bf16 v16bf;
typedef __attribute__((ext_vector_type(8)))  __bf16 v8bf;
typedef __attribute__((ext_vector_type(8)))  float  v8f;

typedef __attribute__((ext_vector_type(4))) unsigned int u32x4;
typedef __attribute__((ext_vector_type(8))) int          i32x8;
typedef __attribute__((ext_vector_type(4))) int          i32x4;

#if defined(__has_builtin)
#if __has_builtin(__builtin_amdgcn_tensor_load_to_lds)
#define HAVE_TDM 1
#endif
#endif

__device__ __forceinline__ float sigf(float x) { return 1.0f / (1.0f + __expf(-x)); }

// Build a 16-element bf16 A-fragment from two 8-element (16B) chunks.
__device__ __forceinline__ v16bf load_frag16(const __bf16* p0, const __bf16* p1) {
  v8bf lo = *(const v8bf*)p0;
  v8bf hi = *(const v8bf*)p1;
  v16bf r;
#pragma unroll
  for (int i = 0; i < 8; ++i) { r[i] = lo[i]; r[i + 8] = hi[i]; }
  return r;
}

#ifdef HAVE_TDM
// TDM: 2-D tile load Global -> LDS, data_size = 2 bytes (bf16).
// Descriptor bit layout per CDNA5 ISA sec. 8.3/8.4.
__device__ __forceinline__ void tdm_load_2d_bf16(unsigned lds_off, const void* gaddr,
                                                 unsigned rows, unsigned cols,
                                                 unsigned stride_elems) {
  unsigned long long ga = (unsigned long long)(uintptr_t)gaddr;
  u32x4 g0;
  g0[0] = 1u;                                            // count=1 (valid), user mode
  g0[1] = lds_off;                                       // lds_addr (bytes)
  g0[2] = (unsigned)(ga & 0xffffffffu);                  // global_addr[31:0]
  g0[3] = (unsigned)((ga >> 32) & 0x01ffffffu) | (2u << 30);  // global_addr[56:32] | type=2
  i32x8 g1;
  g1[0] = (int)(1u << 16);                               // data_size=1 (2 bytes)
  g1[1] = (int)(cols << 16);                             // tensor_dim0[15:0] @ bits[63:48]
  g1[2] = (int)(rows << 16);                             // tensor_dim1[15:0] @ bits[95:80]
  g1[3] = (int)(cols << 16);                             // tile_dim0 @ bits[127:112]
  g1[4] = (int)rows;                                     // tile_dim1 @ bits[143:128]
  g1[5] = (int)stride_elems;                             // tensor_dim0_stride[31:0]
  g1[6] = 0;
  g1[7] = 0;
  i32x4 z4 = {0, 0, 0, 0};
#if __clang_major__ >= 23
  i32x8 z8 = {0, 0, 0, 0, 0, 0, 0, 0};
  __builtin_amdgcn_tensor_load_to_lds(g0, g1, z4, z4, z8, 0);
#else
  __builtin_amdgcn_tensor_load_to_lds(g0, g1, z4, z4, 0);
#endif
}
#endif

// ---------------------------------------------------------------------------
// 1) Embedding gather + concat -> xb bf16, t-major rows: xb[(t*B+b)*D + d]
// ---------------------------------------------------------------------------
__global__ void embed_kernel(const int* __restrict__ word, const int* __restrict__ charr,
                             const float* __restrict__ wemb, const float* __restrict__ cemb,
                             __bf16* __restrict__ xb) {
  int idx = blockIdx.x * blockDim.x + threadIdx.x;  // over T*B*D
  if (idx >= TT * BB * DD) return;
  int d = idx % DD;
  int r = idx / DD;
  int b = r % BB;
  int t = r / BB;
  float v;
  if (d < 128) v = wemb[(size_t)word[b * TT + t] * 128 + d];
  else         v = cemb[(size_t)charr[b * TT + t] * 64 + (d - 128)];
  xb[idx] = (__bf16)v;
}

// ---------------------------------------------------------------------------
// 2) Generic f32 -> bf16 weight conversion
// ---------------------------------------------------------------------------
__global__ void cvt_bf16_kernel(const float* __restrict__ in, __bf16* __restrict__ out, int n) {
  int i = blockIdx.x * blockDim.x + threadIdx.x;
  if (i < n) out[i] = (__bf16)in[i];
}

// ---------------------------------------------------------------------------
// 3) Input projection GEMM (WMMA bf16): xp[dir][(t*B+b)][g] = x.Wih^T + b
//    M=16384, N=512, K=192 (6 k-steps), one 16x16 tile per wave.
// ---------------------------------------------------------------------------
__global__ __launch_bounds__(256) void xproj_kernel(
    const __bf16* __restrict__ xb,    // [T*B][192]
    const __bf16* __restrict__ wih,   // [2][512][192]
    const float* __restrict__ bf, const float* __restrict__ bb,
    float* __restrict__ xp)           // [2][T*B][512]
{
  int w = threadIdx.x >> 5, lane = threadIdx.x & 31;
  int gw = blockIdx.x * 8 + w;        // 65536 waves total
  int dir = gw >> 15;
  int rem = gw & 32767;
  int mt = rem >> 5;                  // 0..1023
  int nt = rem & 31;                  // 0..31
  int ml = lane & 15;
  int hl = (lane & 16) ? 8 : 0;       // A-layout half select
  int bk = (lane & 16) ? 16 : 0;      // unused; B fragment also uses hl
  int col = nt * 16 + ml;

  const __bf16* W = wih + (size_t)dir * G4H * DD;
  const float*  bias = dir ? bb : bf;
  float bv = bias[col];
  v8f acc;
#pragma unroll
  for (int e = 0; e < 8; ++e) acc[e] = bv;

  const __bf16* arow = xb + (size_t)(mt * 16 + ml) * DD;
  const __bf16* brow = W + (size_t)col * DD;
#pragma unroll
  for (int kt = 0; kt < 6; ++kt) {
    int kb = kt * 32 + hl;
    v16bf a  = load_frag16(arow + kb, arow + kb + 16);
    v16bf bm = load_frag16(brow + kb, brow + kb + 16);
    acc = __builtin_amdgcn_wmma_f32_16x16x32_bf16(false, a, false, bm, (short)0, acc, false, false);
  }
  float* orow = xp + (size_t)dir * TT * BB * G4H;
#pragma unroll
  for (int e = 0; e < 8; ++e) {
    int r = mt * 16 + e + hl;
    orow[(size_t)r * G4H + col] = acc[e];
  }
}

// ---------------------------------------------------------------------------
// 4) Persistent BiLSTM scan. 2 blocks (dir), 16 waves each.
//    Wave (m,j): m = batch-tile (0..1), j = hidden-tile (0..7).
//    Owns gate tiles at n = g*128 + j*16 for g=0..3 -> c stays in registers.
//    W_hh (128KB bf16) staged into LDS once via TDM; h (8KB bf16) via LDS.
// ---------------------------------------------------------------------------
__global__ __launch_bounds__(512) void lstm_kernel(
    const float* __restrict__ xp,   // [2][T*B][512] (t-major rows)
    const __bf16* __restrict__ whh, // [2][512][128]
    __bf16* __restrict__ hs)        // [2][T][B][128]
{
  __shared__ __attribute__((aligned(32))) __bf16 w_lds[G4H * HH];  // 128 KB
  __shared__ __attribute__((aligned(16))) __bf16 h_lds[BB * HH];   // 8 KB
  int dir = blockIdx.x;
  int w = threadIdx.x >> 5, lane = threadIdx.x & 31;
  int m = w >> 3;                  // 0..1
  int j = w & 7;                   // 0..7
  int ml = lane & 15;
  int hl = (lane & 16) ? 8 : 0;
  int bk = (lane & 16) ? 16 : 0;

  const float*  xpd = xp  + (size_t)dir * TT * BB * G4H;
  const __bf16* Wd  = whh + (size_t)dir * G4H * HH;
  __bf16*       hsd = hs  + (size_t)dir * TT * BB * HH;

#ifdef HAVE_TDM
  if (w == 0) {  // one TDM descriptor per block moves all of W_hh[dir]
    tdm_load_2d_bf16((unsigned)(uintptr_t)&w_lds[0], Wd, G4H, HH, HH);
    __builtin_amdgcn_s_wait_tensorcnt(0);
  }
#else
  for (int i = threadIdx.x; i < (G4H * HH) / 8; i += 512)
    ((v8bf*)w_lds)[i] = ((const v8bf*)Wd)[i];
#endif
  for (int i = threadIdx.x; i < BB * HH; i += 512) h_lds[i] = (__bf16)0.0f;
  v8f c;
#pragma unroll
  for (int e = 0; e < 8; ++e) c[e] = 0.0f;
  __syncthreads();

  for (int ts = 0; ts < TT; ++ts) {
    int tt = dir ? (TT - 1 - ts) : ts;
    const float* xrow = xpd + (size_t)tt * BB * G4H;
    // warm L2/L0 for next step's xp row while this step computes
    if (ts + 1 < TT) {
      int tn = dir ? (TT - 2 - ts) : (ts + 1);
      __builtin_prefetch(xpd + (size_t)tn * BB * G4H + (size_t)(m * 16 + hl) * G4H + j * 16 + ml, 0, 1);
    }
    v8f acc[4];
#pragma unroll
    for (int g = 0; g < 4; ++g) {
      int col = g * 128 + j * 16 + ml;
#pragma unroll
      for (int e = 0; e < 8; ++e)
        acc[g][e] = xrow[(size_t)(m * 16 + e + hl) * G4H + col];
    }
    const __bf16* hrow = &h_lds[(m * 16 + ml) * HH];
#pragma unroll
    for (int kt = 0; kt < 4; ++kt) {
      int kb = kt * 32 + hl;
      v16bf a = load_frag16(hrow + kb, hrow + kb + 16);
#pragma unroll
      for (int g = 0; g < 4; ++g) {
        int col = g * 128 + j * 16 + ml;
        v16bf bm = load_frag16(&w_lds[(size_t)col * HH + kb], &w_lds[(size_t)col * HH + kb + 16]);
        acc[g] = __builtin_amdgcn_wmma_f32_16x16x32_bf16(false, a, false, bm, (short)0, acc[g], false, false);
      }
    }
    __syncthreads();  // all reads of h(t-1) done
#pragma unroll
    for (int e = 0; e < 8; ++e) {
      float ig = sigf(acc[0][e]);
      float fg = sigf(acc[1][e]);
      float gg = tanhf(acc[2][e]);
      float og = sigf(acc[3][e]);
      float cn = fg * c[e] + ig * gg;
      c[e] = cn;
      float hv = og * tanhf(cn);
      int row  = m * 16 + e + hl;     // batch index
      int colh = j * 16 + ml;         // hidden index
      __bf16 hb = (__bf16)hv;
      h_lds[row * HH + colh] = hb;
      hsd[((size_t)tt * BB + row) * HH + colh] = hb;
    }
    __syncthreads();  // h(t) visible
  }
}

// ---------------------------------------------------------------------------
// 5) Output projection (WMMA bf16): emit[b][t][l] = [h_f|h_b].Wout^T + b, masked
//    M=16384 (rows r=t*32+b), N=32, K=256 (8 k-steps: 0-3 fwd, 4-7 bwd)
// ---------------------------------------------------------------------------
__global__ __launch_bounds__(256) void emit_kernel(
    const __bf16* __restrict__ hs,    // [2][T][B][128]
    const __bf16* __restrict__ woutb, // [32][256]
    const float* __restrict__ b_out,
    const int* __restrict__ charr,
    float* __restrict__ emit)         // [B][T][32]
{
  int w = threadIdx.x >> 5, lane = threadIdx.x & 31;
  int gw = blockIdx.x * 8 + w;       // 2048 waves
  int mt = gw >> 1;
  int nt = gw & 1;
  int ml = lane & 15;
  int hl = (lane & 16) ? 8 : 0;
  int bk = (lane & 16) ? 16 : 0;
  int col = nt * 16 + ml;            // label l

  v8f acc;
  float bv = b_out[col];
#pragma unroll
  for (int e = 0; e < 8; ++e) acc[e] = bv;

  int rowA = mt * 16 + ml;           // r = t*32+b
  int tA = rowA >> 5, bA = rowA & 31;
#pragma unroll
  for (int kt = 0; kt < 8; ++kt) {
    int ds = kt >> 2;
    int kb = (kt & 3) * 32 + hl;
    const __bf16* ap = hs + (((size_t)ds * TT + tA) * BB + bA) * HH + kb;
    v16bf a  = load_frag16(ap, ap + 16);
    v16bf bm = load_frag16(woutb + (size_t)col * 256 + kt * 32 + hl, woutb + (size_t)col * 256 + kt * 32 + hl + 16);
    acc = __builtin_amdgcn_wmma_f32_16x16x32_bf16(false, a, false, bm, (short)0, acc, false, false);
  }
#pragma unroll
  for (int e = 0; e < 8; ++e) {
    int r = mt * 16 + e + hl;
    int t = r >> 5, b = r & 31;
    float mask = (charr[b * TT + t] > 0) ? 1.0f : 0.0f;
    emit[((size_t)b * TT + t) * LL + col] = acc[e] * mask;
  }
}

// ---------------------------------------------------------------------------
// 6) CRF: forward algorithm + gold score. 1 block, 32 waves (wave = batch),
//    32 lanes = 32 labels. Online logsumexp via __shfl broadcast.
// ---------------------------------------------------------------------------
__global__ __launch_bounds__(1024) void crf_kernel(
    const float* __restrict__ emit, const float* __restrict__ trans,
    const float* __restrict__ startv, const float* __restrict__ endv,
    const int* __restrict__ charr, const int* __restrict__ y,
    float* __restrict__ nll)
{
  __shared__ float tr[LL * LL];
  for (int i = threadIdx.x; i < LL * LL; i += 1024) tr[i] = trans[i];
  __syncthreads();

  int b = threadIdx.x >> 5, j = threadIdx.x & 31;
  const float* eb = emit + (size_t)b * TT * LL;
  const int* cb = charr + b * TT;
  const int* yb = y + b * TT;

  float alpha = startv[j] + eb[j];  // t = 0
  for (int t = 1; t < TT; ++t) {
    float m = -__builtin_inff(), s = 0.0f;
    for (int i = 0; i < LL; ++i) {
      float ai = __shfl(alpha, i, 32);
      float v = ai + tr[i * LL + j];
      if (v > m) { s = s * __expf(m - v) + 1.0f; m = v; }
      else       { s += __expf(v - m); }
    }
    float nv = m + __logf(s) + eb[t * LL + j];
    if (cb[t] > 0) alpha = nv;
  }
  // logZ = logsumexp_j(alpha + end)
  float v = alpha + endv[j];
  float m = v;
#pragma unroll
  for (int off = 16; off; off >>= 1) m = fmaxf(m, __shfl_xor(m, off, 32));
  float s = __expf(v - m);
#pragma unroll
  for (int off = 16; off; off >>= 1) s += __shfl_xor(s, off, 32);
  float logZ = m + __logf(s);

  // gold path score (lane-parallel over t)
  float es = 0.0f, tsc = 0.0f;
  int cnt = 0;
  for (int t = j; t < TT; t += 32) {
    float mk = (cb[t] > 0) ? 1.0f : 0.0f;
    es += eb[t * LL + yb[t]] * mk;
    cnt += (cb[t] > 0) ? 1 : 0;
  }
  for (int t = 1 + j; t < TT; t += 32) {
    float mk = (cb[t] > 0) ? 1.0f : 0.0f;
    tsc += tr[yb[t - 1] * LL + yb[t]] * mk;
  }
#pragma unroll
  for (int off = 16; off; off >>= 1) {
    es  += __shfl_xor(es, off, 32);
    tsc += __shfl_xor(tsc, off, 32);
    cnt += __shfl_xor(cnt, off, 32);
  }
  if (j == 0) {
    int last = cnt - 1;
    float gold = startv[yb[0]] + es + tsc + endv[yb[last]];
    nll[b] = logZ - gold;
  }
}

// 7) Deterministic final reduction
__global__ void reduce_kernel(const float* __restrict__ nll, float* __restrict__ out) {
  if (threadIdx.x == 0 && blockIdx.x == 0) {
    float s = 0.0f;
    for (int i = 0; i < BB; ++i) s += nll[i];
    out[0] = s;
  }
}

// ---------------------------------------------------------------------------
extern "C" void kernel_launch(void* const* d_in, const int* in_sizes, int n_in,
                              void* d_out, int out_size, void* d_ws, size_t ws_size,
                              hipStream_t stream) {
  const int*   word    = (const int*)d_in[0];
  const int*   charr   = (const int*)d_in[1];
  const int*   y       = (const int*)d_in[2];
  const float* wordemb = (const float*)d_in[3];
  const float* charemb = (const float*)d_in[4];
  const float* W_ih_f  = (const float*)d_in[5];
  const float* W_hh_f  = (const float*)d_in[6];
  const float* b_f     = (const float*)d_in[7];
  const float* W_ih_b  = (const float*)d_in[8];
  const float* W_hh_b  = (const float*)d_in[9];
  const float* b_b     = (const float*)d_in[10];
  const float* W_out   = (const float*)d_in[11];
  const float* b_out   = (const float*)d_in[12];
  const float* trans   = (const float*)d_in[13];
  const float* startv  = (const float*)d_in[14];
  const float* endv    = (const float*)d_in[15];
  float* out = (float*)d_out;

  uint8_t* ws = (uint8_t*)d_ws;
  size_t o = 0;
  auto carve = [&](size_t bytes) -> void* {
    void* p = ws + o;
    o = (o + bytes + 255) & ~(size_t)255;
    return p;
  };
  __bf16* xb    = (__bf16*)carve((size_t)TT * BB * DD * 2);         // 6 MB
  __bf16* wih2  = (__bf16*)carve((size_t)2 * G4H * DD * 2);         // 384 KB
  __bf16* whh2  = (__bf16*)carve((size_t)2 * G4H * HH * 2);         // 256 KB
  __bf16* woutb = (__bf16*)carve((size_t)LL * 2 * HH * 2);          // 16 KB
  float*  xp    = (float*)carve((size_t)2 * TT * BB * G4H * 4);     // 64 MB
  __bf16* hsbuf = (__bf16*)carve((size_t)2 * TT * BB * HH * 2);     // 8 MB
  float*  emit  = (float*)carve((size_t)BB * TT * LL * 4);          // 2 MB
  float*  nll   = (float*)carve((size_t)BB * 4);

  // 1) embeddings -> bf16 t-major x
  {
    int n = TT * BB * DD;
    embed_kernel<<<(n + 255) / 256, 256, 0, stream>>>(word, charr, wordemb, charemb, xb);
  }
  // 2) weights -> bf16
  {
    int n = G4H * DD;
    cvt_bf16_kernel<<<(n + 255) / 256, 256, 0, stream>>>(W_ih_f, wih2, n);
    cvt_bf16_kernel<<<(n + 255) / 256, 256, 0, stream>>>(W_ih_b, wih2 + n, n);
    n = G4H * HH;
    cvt_bf16_kernel<<<(n + 255) / 256, 256, 0, stream>>>(W_hh_f, whh2, n);
    cvt_bf16_kernel<<<(n + 255) / 256, 256, 0, stream>>>(W_hh_b, whh2 + n, n);
    n = LL * 2 * HH;
    cvt_bf16_kernel<<<(n + 255) / 256, 256, 0, stream>>>(W_out, woutb, n);
  }
  // 3) input projections: 65536 waves = 8192 blocks x 8 waves
  xproj_kernel<<<8192, 256, 0, stream>>>(xb, wih2, b_f, b_b, xp);
  // 4) persistent BiLSTM: one block per direction
  lstm_kernel<<<2, 512, 0, stream>>>(xp, whh2, hsbuf);
  // 5) emissions: 2048 waves = 256 blocks x 8 waves
  emit_kernel<<<256, 256, 0, stream>>>(hsbuf, woutb, b_out, charr, emit);
  // 6) CRF forward + gold
  crf_kernel<<<1, 1024, 0, stream>>>(emit, trans, startv, endv, charr, y, nll);
  // 7) deterministic sum
  reduce_kernel<<<1, 32, 0, stream>>>(nll, out);
}
